// GCNN_44916767981906
// MI455X (gfx1250) — matrix operand order repaired and checked by reading, not compiled, not run on hardware
//
#include <hip/hip_runtime.h>

#define NN 8192
#define FF 133
#define HH 256
#define MF 200
#define RH 512
#define DEPTH 3
#define N_RO 2

typedef __attribute__((ext_vector_type(16))) _Float16 v16h;
typedef __attribute__((ext_vector_type(8)))  _Float16 v8h;
typedef __attribute__((ext_vector_type(8)))  float    v8f;
typedef __attribute__((ext_vector_type(4)))  float    v4f;

// sigmoid via hardware transcendentals: v_exp_f32 (base-2) + v_rcp_f32
__device__ __forceinline__ float sigm(float x) {
    const float e = __builtin_amdgcn_exp2f(-1.44269504088896341f * x);
    return __builtin_amdgcn_rcpf(1.0f + e);
}

// ---------------------------------------------------------------------------
// h = X @ W_in + b_in      (8192x133) @ (133x256)
// 8 rows per block, 256 threads (one per output column)
// ---------------------------------------------------------------------------
__global__ __launch_bounds__(256) void k_in(const float* __restrict__ X,
                                            const float* __restrict__ W,
                                            const float* __restrict__ b,
                                            float* __restrict__ h) {
    __shared__ float xs[8 * FF];
    const int r0 = blockIdx.x * 8;
    for (int idx = threadIdx.x; idx < 8 * FF; idx += 256)
        xs[idx] = X[(size_t)r0 * FF + idx];
    __syncthreads();
    const int c = threadIdx.x;
    const float bias = b[c];
    float acc[8];
#pragma unroll
    for (int r = 0; r < 8; ++r) acc[r] = bias;
    for (int k = 0; k < FF; ++k) {
        const float w = W[k * HH + c];
#pragma unroll
        for (int r = 0; r < 8; ++r) acc[r] += xs[r * FF + k] * w;
    }
#pragma unroll
    for (int r = 0; r < 8; ++r) h[(size_t)(r0 + r) * HH + c] = acc[r];
}

// ---------------------------------------------------------------------------
// s_nei = h @ W_att[:H], s_self = h @ W_att[H:], and hT = f16 transpose of h.
// One wave per row; 8 rows per 256-thread block.
// ---------------------------------------------------------------------------
__global__ __launch_bounds__(256) void k_scores(const float* __restrict__ h,
                                                const float* __restrict__ Watt,
                                                float* __restrict__ s_self,
                                                float* __restrict__ s_nei,
                                                _Float16* __restrict__ hT) {
    const int lane = threadIdx.x & 31;
    const int wv   = threadIdx.x >> 5;
    const int row  = blockIdx.x * 8 + wv;
    const float* hr = h + (size_t)row * HH;
    const int base = lane * 8;
    float pn = 0.f, ps = 0.f;
    float v[8];
#pragma unroll
    for (int e = 0; e < 8; ++e) {
        const float x = hr[base + e];
        v[e] = x;
        pn += x * Watt[base + e];
        ps += x * Watt[HH + base + e];
    }
#pragma unroll
    for (int e = 0; e < 8; ++e)
        hT[(size_t)(base + e) * NN + row] = (_Float16)v[e];
#pragma unroll
    for (int off = 16; off > 0; off >>= 1) {
        pn += __shfl_xor(pn, off, 32);
        ps += __shfl_xor(ps, off, 32);
    }
    if (lane == 0) { s_nei[row] = pn; s_self[row] = ps; }
}

// ---------------------------------------------------------------------------
// h2 = (A ⊙ sigmoid(s_self[i] + s_nei[j] + b_att)) @ h
// WMMA f32<-f16 GEMM, A-fragment fused (mask * sigmoid) on the fly.
// Block: 256 threads = 8 waves. Waves (2t,2t+1) share a 16-row M-tile and
// split the 256 output columns into two 128-col halves (8 N-tiles each).
// Grid: NN/64 = 128 blocks.
// Per K-tile: issue all 16 B-fragment b128 loads first, overlap their L2
// latency with the sigmoid/afrag VALU work, then drain 8 back-to-back WMMAs.
// Fragment layouts per CDNA5 ISA §7.12.2:
//   A 16x32 f16 : lane m=lane&15, half=lane>>4; elem e<8 -> k=half*8+e,
//                 e>=8 -> k=16+half*8+(e-8)
//   B 32x16 f16 : col n=lane&15; same k order as A (two 8-runs in hT)
//   C/D 16x16 f32: row = half*8 + v, col = lane&15
// ---------------------------------------------------------------------------
__global__ __launch_bounds__(256) void k_att(const float* __restrict__ adj,
                                             const float* __restrict__ s_self,
                                             const float* __restrict__ s_nei,
                                             const float* __restrict__ b_att_p,
                                             const _Float16* __restrict__ hT,
                                             float* __restrict__ out) {
    const float batt = b_att_p[0];
    const int lane = threadIdx.x & 31;
    const int wv   = threadIdx.x >> 5;
    const int mt   = wv >> 1;          // 0..3: M-tile within block
    const int nh   = wv & 1;           // 0..1: which half of the columns
    const int i0   = blockIdx.x * 64 + mt * 16;
    const int ntB  = nh * 8;           // first N-tile
    const int m    = lane & 15;
    const int hx   = lane >> 4;        // lane half
    const int row  = i0 + m;

    const float* arow = adj + (size_t)row * NN;
    // base pointer for this lane's B-fragment column group; per-nt offsets are
    // compile-time immediates (nt * 16 columns * NN halves)
    const _Float16* bbase = hT + (size_t)(ntB * 16 + m) * NN + hx * 8;
    const float sself = s_self[row] + batt;

    v8f acc[8];
#pragma unroll
    for (int nt = 0; nt < 8; ++nt)
        acc[nt] = (v8f){0.f, 0.f, 0.f, 0.f, 0.f, 0.f, 0.f, 0.f};

    for (int kb = 0; kb < NN; kb += 32) {
        const int j0 = kb + hx * 8;
        const int j1 = kb + 16 + hx * 8;
        const v4f a0 = *(const v4f*)(arow + j0);
        const v4f a1 = *(const v4f*)(arow + j0 + 4);
        const v4f a2 = *(const v4f*)(arow + j1);
        const v4f a3 = *(const v4f*)(arow + j1 + 4);

        // adjacency entries are {0,1}: skip tile if whole wave sees zeros
        const float t = a0[0]+a0[1]+a0[2]+a0[3] + a1[0]+a1[1]+a1[2]+a1[3]
                      + a2[0]+a2[1]+a2[2]+a2[3] + a3[0]+a3[1]+a3[2]+a3[3];
        if (__ballot(t != 0.f) == 0ull) continue;

        // kick off all B-fragment loads (2x b128 each) before the VALU work
        v8h blo[8], bhi[8];
#pragma unroll
        for (int nt = 0; nt < 8; ++nt) {
            const _Float16* bp = bbase + (size_t)nt * 16 * NN + kb;
            blo[nt] = *(const v8h*)bp;
            bhi[nt] = *(const v8h*)(bp + 16);
        }

        const v4f n0 = *(const v4f*)(s_nei + j0);
        const v4f n1 = *(const v4f*)(s_nei + j0 + 4);
        const v4f n2 = *(const v4f*)(s_nei + j1);
        const v4f n3 = *(const v4f*)(s_nei + j1 + 4);

        v16h afrag;
#pragma unroll
        for (int e = 0; e < 4; ++e) {
            afrag[e]      = (_Float16)(a0[e] * sigm(sself + n0[e]));
            afrag[4 + e]  = (_Float16)(a1[e] * sigm(sself + n1[e]));
            afrag[8 + e]  = (_Float16)(a2[e] * sigm(sself + n2[e]));
            afrag[12 + e] = (_Float16)(a3[e] * sigm(sself + n3[e]));
        }

#pragma unroll
        for (int nt = 0; nt < 8; ++nt) {
            const v16h bfrag = __builtin_shufflevector(
                blo[nt], bhi[nt],
                0, 1, 2, 3, 4, 5, 6, 7, 8, 9, 10, 11, 12, 13, 14, 15);
            acc[nt] = __builtin_amdgcn_wmma_f32_16x16x32_f16(
                false, afrag, false, bfrag, (short)0, acc[nt], false, false);
        }
    }

#pragma unroll
    for (int nt = 0; nt < 8; ++nt) {
        const int col = (ntB + nt) * 16 + m;
#pragma unroll
        for (int v = 0; v < 8; ++v) {
            const int r = i0 + hx * 8 + v;
            out[(size_t)r * HH + col] = acc[nt][v];
        }
    }
}

// ---------------------------------------------------------------------------
// h = relu(h2 @ W_node[d] + b_node[d])   (8192x256)@(256x256)
// ---------------------------------------------------------------------------
__global__ __launch_bounds__(256) void k_node(const float* __restrict__ x,
                                              const float* __restrict__ W,
                                              const float* __restrict__ b,
                                              float* __restrict__ y) {
    __shared__ float xs[8 * HH];
    const int r0 = blockIdx.x * 8;
    for (int idx = threadIdx.x; idx < 8 * HH; idx += 256)
        xs[idx] = x[(size_t)r0 * HH + idx];
    __syncthreads();
    const int c = threadIdx.x;
    const float bias = b[c];
    float acc[8];
#pragma unroll
    for (int r = 0; r < 8; ++r) acc[r] = bias;
    for (int k = 0; k < HH; ++k) {
        const float w = W[k * HH + c];
#pragma unroll
        for (int r = 0; r < 8; ++r) acc[r] += xs[r * HH + k] * w;
    }
#pragma unroll
    for (int r = 0; r < 8; ++r)
        y[(size_t)(r0 + r) * HH + c] = fmaxf(acc[r], 0.f);
}

// ---------------------------------------------------------------------------
// Column sum of h (axis 0), two stages.
// ---------------------------------------------------------------------------
__global__ __launch_bounds__(256) void k_colsum_part(const float* __restrict__ h,
                                                     float* __restrict__ partial) {
    const int c = threadIdx.x;
    const int base = blockIdx.x * 128;
    float a0 = 0.f, a1 = 0.f, a2 = 0.f, a3 = 0.f;
    for (int i = 0; i < 128; i += 4) {
        a0 += h[(size_t)(base + i + 0) * HH + c];
        a1 += h[(size_t)(base + i + 1) * HH + c];
        a2 += h[(size_t)(base + i + 2) * HH + c];
        a3 += h[(size_t)(base + i + 3) * HH + c];
    }
    partial[blockIdx.x * HH + c] = (a0 + a1) + (a2 + a3);
}

__global__ __launch_bounds__(256) void k_colsum_fin(const float* __restrict__ partial,
                                                    const float* __restrict__ mol,
                                                    float* __restrict__ g_ext) {
    const int c = threadIdx.x;
    float a = 0.f;
    for (int b = 0; b < 64; ++b) a += partial[b * HH + c];
    g_ext[c] = a;
    if (c < MF) g_ext[HH + c] = mol[c];
}

// ---------------------------------------------------------------------------
// Readout MLP: 456 -> 512 (relu) -> 512 (relu) -> 512 (relu) -> scalar
// Single block, 512 threads.
// ---------------------------------------------------------------------------
__global__ __launch_bounds__(512) void k_readout(const float* __restrict__ g_ext,
                                                 const float* __restrict__ W_ro_in,
                                                 const float* __restrict__ b_ro_in,
                                                 const float* __restrict__ W_ro_hid,
                                                 const float* __restrict__ b_ro_hid,
                                                 const float* __restrict__ W_out,
                                                 const float* __restrict__ b_out,
                                                 float* __restrict__ out) {
    __shared__ float sg[HH + MF];
    __shared__ float b0[RH];
    __shared__ float b1[RH];
    const int o = threadIdx.x;
    if (o < HH + MF) sg[o] = g_ext[o];
    __syncthreads();

    float acc = b_ro_in[o];
    for (int k = 0; k < HH + MF; ++k) acc += sg[k] * W_ro_in[k * RH + o];
    b0[o] = fmaxf(acc, 0.f);

    float* src = b0;
    float* dst = b1;
    for (int d = 0; d < N_RO; ++d) {
        __syncthreads();
        float a = b_ro_hid[d * RH + o];
        const float* Wd = W_ro_hid + (size_t)d * RH * RH;
        for (int k = 0; k < RH; ++k) a += src[k] * Wd[k * RH + o];
        dst[o] = fmaxf(a, 0.f);
        float* tmp = src; src = dst; dst = tmp;
    }
    __syncthreads();
    dst[o] = src[o] * W_out[o];     // dst is free scratch now
    __syncthreads();
    for (int s = 256; s > 0; s >>= 1) {
        if (o < s) dst[o] += dst[o + s];
        __syncthreads();
    }
    if (o == 0) out[0] = dst[0] + b_out[0];
}

// ---------------------------------------------------------------------------
extern "C" void kernel_launch(void* const* d_in, const int* in_sizes, int n_in,
                              void* d_out, int out_size, void* d_ws, size_t ws_size,
                              hipStream_t stream) {
    const float* adj      = (const float*)d_in[0];
    const float* X        = (const float*)d_in[1];
    const float* mol      = (const float*)d_in[2];
    const float* W_in     = (const float*)d_in[3];
    const float* b_in     = (const float*)d_in[4];
    const float* W_att    = (const float*)d_in[5];
    const float* b_att    = (const float*)d_in[6];
    const float* W_node   = (const float*)d_in[7];
    const float* b_node   = (const float*)d_in[8];
    const float* W_ro_in  = (const float*)d_in[9];
    const float* b_ro_in  = (const float*)d_in[10];
    const float* W_ro_hid = (const float*)d_in[11];
    const float* b_ro_hid = (const float*)d_in[12];
    const float* W_out    = (const float*)d_in[13];
    const float* b_out    = (const float*)d_in[14];
    float* out = (float*)d_out;

    char* ws = (char*)d_ws;
    float*    h       = (float*)(ws);                                  //  8 MiB
    float*    h2      = (float*)(ws + (size_t)8  * 1024 * 1024);       //  8 MiB
    _Float16* hT      = (_Float16*)(ws + (size_t)16 * 1024 * 1024);    //  4 MiB
    float*    s_self  = (float*)(ws + (size_t)20 * 1024 * 1024);
    float*    s_nei   = (float*)(ws + (size_t)20 * 1024 * 1024 + 64 * 1024);
    float*    partial = (float*)(ws + (size_t)20 * 1024 * 1024 + 128 * 1024);
    float*    g_ext   = (float*)(ws + (size_t)20 * 1024 * 1024 + 256 * 1024);

    k_in<<<NN / 8, 256, 0, stream>>>(X, W_in, b_in, h);
    for (int d = 0; d < DEPTH; ++d) {
        k_scores<<<NN / 8, 256, 0, stream>>>(h, W_att, s_self, s_nei, hT);
        k_att<<<NN / 64, 256, 0, stream>>>(adj, s_self, s_nei, b_att, hT, h2);
        k_node<<<NN / 8, 256, 0, stream>>>(h2, W_node + (size_t)d * HH * HH,
                                           b_node + (size_t)d * HH, h);
    }
    k_colsum_part<<<64, 256, 0, stream>>>(h, partial);
    k_colsum_fin<<<1, 256, 0, stream>>>(partial, mol, g_ext);
    k_readout<<<1, 512, 0, stream>>>(g_ext, W_ro_in, b_ro_in, W_ro_hid,
                                     b_ro_hid, W_out, b_out, out);
}
